// VarianceAdaptor_46368466927999
// MI455X (gfx1250) — compile-verified, matrix operand order checked by reading, not run on hardware
//
#include <hip/hip_runtime.h>
#include <hip/hip_bf16.h>

typedef _Float16 half_t;
typedef _Float16 v16h __attribute__((ext_vector_type(16)));
typedef _Float16 v8h  __attribute__((ext_vector_type(8)));
typedef float    v8f  __attribute__((ext_vector_type(8)));

#define BB   32
#define SS   512
#define TT   2048
#define DD   256
#define KDIM 768
#define NKT  24            // KDIM / 32
#define CHUNK 8            // batch items per h1 chunk
#define MT   32            // rows per block tile
#define LDSA_STRIDE 264    // 256 + 8 halves (16B aligned rows, conflict padding)
#define LDSO_STRIDE 260

// ---------------------------------------------------------------------------
// Pack conv weight [256(co), Cin=256, 3] fp32 -> f16 B-fragment layout:
//   out[((kt*4 + q)*256 + n)*8 + j] = w[n, ci, k]  with kt*32 + q*8 + j = k*256 + ci
// so a lane's 16-half B fragment is two contiguous 16B chunks (q=s and q=s+2).
// ---------------------------------------------------------------------------
__global__ void pack_w_kernel(const float* __restrict__ w, half_t* __restrict__ out) {
    int tid = blockIdx.x * blockDim.x + threadIdx.x;   // 0 .. 768*256-1
    int j  = tid & 7;
    int n  = (tid >> 3) & 255;
    int q  = (tid >> 11) & 3;
    int kt = tid >> 13;
    int kkk = kt * 32 + q * 8 + j;      // K index = k*256 + ci
    int k  = kkk >> 8;
    int ci = kkk & 255;
    out[tid] = (half_t)w[n * (DD * 3) + ci * 3 + k];
}

// ---------------------------------------------------------------------------
// Per-batch cumsum of duration targets + mel_len output (as float)
// ---------------------------------------------------------------------------
__global__ void cumsum_kernel(const int* __restrict__ dur, int* __restrict__ cum,
                              float* __restrict__ mel_len) {
    int b = threadIdx.x;
    if (b < BB) {
        int acc = 0;
        for (int i = 0; i < SS; ++i) { acc += dur[b * SS + i]; cum[b * SS + i] = acc; }
        mel_len[b] = (float)acc;
    }
}

// ---------------------------------------------------------------------------
// Length regulation: xe[b,t,:] = x[b, searchsorted_right(cum,t), :] * mask
// One wave per t (8 t per block of 256 threads).
// ---------------------------------------------------------------------------
__global__ void regulate_kernel(const float* __restrict__ x, const int* __restrict__ cum,
                                float* __restrict__ xe) {
    int b = blockIdx.y;
    int t = blockIdx.x * 8 + (threadIdx.x >> 5);
    int lane = threadIdx.x & 31;
    const int* c = cum + b * SS;
    int lo = 0, hi = SS;
    while (lo < hi) { int mid = (lo + hi) >> 1; if (c[mid] <= t) lo = mid + 1; else hi = mid; }
    int idx = lo < (SS - 1) ? lo : (SS - 1);
    bool mask = t < c[SS - 1];
    const float* src = x + ((size_t)b * SS + idx) * DD;
    float* dst = xe + ((size_t)b * TT + t) * DD;
    #pragma unroll
    for (int j = 0; j < 8; ++j) {
        int ch = lane + j * 32;
        dst[ch] = mask ? src[ch] : 0.0f;
    }
}

// ---------------------------------------------------------------------------
// xe[b,t,:] += emb[searchsorted_left(bins, target[b,t]), :]
// ---------------------------------------------------------------------------
__global__ void addemb_kernel(const float* __restrict__ target, const float* __restrict__ bins,
                              const float* __restrict__ emb, float* __restrict__ xe) {
    int b = blockIdx.y;
    int t = blockIdx.x * 8 + (threadIdx.x >> 5);
    int lane = threadIdx.x & 31;
    float v = target[(size_t)b * TT + t];
    int lo = 0, hi = 255;                 // NBINS-1 boundaries
    while (lo < hi) { int mid = (lo + hi) >> 1; if (bins[mid] < v) lo = mid + 1; else hi = mid; }
    const float* e = emb + (size_t)lo * DD;
    float* dst = xe + ((size_t)b * TT + t) * DD;
    #pragma unroll
    for (int j = 0; j < 8; ++j) {
        int ch = lane + j * 32;
        dst[ch] += e[ch];
    }
}

// ---------------------------------------------------------------------------
// Fused Conv1d(K=3) GEMM (WMMA f16 -> f32) + bias + LayerNorm + ReLU,
// then either store h (f16) or fuse Linear(256->1) into a per-row scalar.
// Block: 256 threads (8 waves), tile = 32 rows x 256 cols, K loop = 24 x 32.
// Each wave: 2 M-subtiles x 2 N-tiles -> 4 WMMAs per B-fragment pair
// (halves L2 weight traffic vs a 16-row tile).
// ---------------------------------------------------------------------------
template <typename TIN, bool FINAL>
__global__ __launch_bounds__(256) void conv_kernel(
    const TIN* __restrict__ in,        // [nb][L][256], chunk base pre-applied
    const half_t* __restrict__ wp,     // packed weights [24][4][256][8] halves
    const float* __restrict__ bias,
    const float* __restrict__ gamma, const float* __restrict__ beta,
    const float* __restrict__ lw, const float* __restrict__ lb,   // FINAL only
    half_t* __restrict__ hout,         // !FINAL: [nb][L][256] f16
    float* __restrict__ pred,          // FINAL: pred[by*L + t] (chunk base applied)
    int L) {
    __shared__ __align__(16) half_t sA[(MT + 2) * LDSA_STRIDE];
    __shared__ float sOut[MT * LDSO_STRIDE];

    const int tid = threadIdx.x;
    const int by = blockIdx.y;
    const int t0 = blockIdx.x * MT;
    const TIN* inb = in + (size_t)by * L * DD;

    // Stage input rows t0-1 .. t0+MT into LDS as f16 (zero-pad at edges).
    for (int r = 0; r < MT + 2; ++r) {
        int t = t0 + r - 1;
        float v = 0.0f;
        if (t >= 0 && t < L) v = (float)inb[(size_t)t * DD + tid];
        sA[r * LDSA_STRIDE + tid] = (half_t)v;
    }
    __syncthreads();

    const int wave = tid >> 5;
    const int lane = tid & 31;
    const int m = lane & 15;       // row (A) / col-within-tile (B)
    const int s = lane >> 4;       // lane-half: selects K groups
    const int n0 = wave * 32;      // this wave's 2 N-tiles: [n0, n0+16) and [n0+16, n0+32)

    v8f c00 = {0.f, 0.f, 0.f, 0.f, 0.f, 0.f, 0.f, 0.f};
    v8f c01 = c00, c10 = c00, c11 = c00;

    for (int kt = 0; kt < NKT; ++kt) {
        const int kk0 = kt * 32;
        const int tap = kk0 >> 8;          // conv tap (0..2)
        const int ci0 = kk0 & 255;         // channel base within tap
        // A fragments: halves k = s*8 + {0..7} and s*8 + 16 + {0..7}
        const half_t* arow0 = &sA[(m + tap) * LDSA_STRIDE + ci0 + s * 8];
        const half_t* arow1 = arow0 + 16 * LDSA_STRIDE;
        v8h a0lo = *(const v8h*)arow0;
        v8h a0hi = *(const v8h*)(arow0 + 16);
        v8h a1lo = *(const v8h*)arow1;
        v8h a1hi = *(const v8h*)(arow1 + 16);
        v16h a0 = __builtin_shufflevector(a0lo, a0hi,
                                          0, 1, 2, 3, 4, 5, 6, 7,
                                          8, 9, 10, 11, 12, 13, 14, 15);
        v16h a1 = __builtin_shufflevector(a1lo, a1hi,
                                          0, 1, 2, 3, 4, 5, 6, 7,
                                          8, 9, 10, 11, 12, 13, 14, 15);
        const half_t* wq0 = wp + ((size_t)(kt * 4 + s) * 256) * 8;
        const half_t* wq2 = wp + ((size_t)(kt * 4 + s + 2) * 256) * 8;
        v8h b0lo = *(const v8h*)(wq0 + (size_t)(n0 + m) * 8);
        v8h b0hi = *(const v8h*)(wq2 + (size_t)(n0 + m) * 8);
        v8h b1lo = *(const v8h*)(wq0 + (size_t)(n0 + 16 + m) * 8);
        v8h b1hi = *(const v8h*)(wq2 + (size_t)(n0 + 16 + m) * 8);
        v16h b0 = __builtin_shufflevector(b0lo, b0hi,
                                          0, 1, 2, 3, 4, 5, 6, 7,
                                          8, 9, 10, 11, 12, 13, 14, 15);
        v16h b1 = __builtin_shufflevector(b1lo, b1hi,
                                          0, 1, 2, 3, 4, 5, 6, 7,
                                          8, 9, 10, 11, 12, 13, 14, 15);
        c00 = __builtin_amdgcn_wmma_f32_16x16x32_f16(false, a0, false, b0, (short)0, c00, false, false);
        c01 = __builtin_amdgcn_wmma_f32_16x16x32_f16(false, a0, false, b1, (short)0, c01, false, false);
        c10 = __builtin_amdgcn_wmma_f32_16x16x32_f16(false, a1, false, b0, (short)0, c10, false, false);
        c11 = __builtin_amdgcn_wmma_f32_16x16x32_f16(false, a1, false, b1, (short)0, c11, false, false);
    }

    // Scatter C (+conv bias) to LDS. C layout: VGPR v -> M = v + 8*s, N = lane&15.
    {
        float bv0 = bias[n0 + m];
        float bv1 = bias[n0 + 16 + m];
        #pragma unroll
        for (int v = 0; v < 8; ++v) {
            int row = v + 8 * s;
            sOut[row * LDSO_STRIDE + n0 + m]             = c00[v] + bv0;
            sOut[row * LDSO_STRIDE + n0 + 16 + m]        = c01[v] + bv1;
            sOut[(row + 16) * LDSO_STRIDE + n0 + m]      = c10[v] + bv0;
            sOut[(row + 16) * LDSO_STRIDE + n0 + 16 + m] = c11[v] + bv1;
        }
    }
    __syncthreads();

    // LayerNorm (+ReLU, + optional fused Linear) — 4 rows per wave.
    for (int rr = 0; rr < 4; ++rr) {
        int r = wave * 4 + rr;
        float s1 = 0.f, s2 = 0.f;
        #pragma unroll
        for (int j = 0; j < 8; ++j) {
            float xv = sOut[r * LDSO_STRIDE + lane + j * 32];
            s1 += xv; s2 += xv * xv;
        }
        #pragma unroll
        for (int off = 16; off >= 1; off >>= 1) {
            s1 += __shfl_xor(s1, off, 32);
            s2 += __shfl_xor(s2, off, 32);
        }
        float mean = s1 * (1.0f / 256.0f);
        float var  = s2 * (1.0f / 256.0f) - mean * mean;
        float rstd = rsqrtf(var + 1e-5f);
        if constexpr (FINAL) {
            float acc = 0.f;
            #pragma unroll
            for (int j = 0; j < 8; ++j) {
                int ch = lane + j * 32;
                float xv = sOut[r * LDSO_STRIDE + ch];
                float y = (xv - mean) * rstd * gamma[ch] + beta[ch];
                y = fmaxf(y, 0.0f);
                acc += y * lw[ch];
            }
            #pragma unroll
            for (int off = 16; off >= 1; off >>= 1) acc += __shfl_xor(acc, off, 32);
            if (lane == 0) pred[(size_t)by * L + t0 + r] = acc + lb[0];
        } else {
            #pragma unroll
            for (int j = 0; j < 8; ++j) {
                int ch = lane + j * 32;
                float xv = sOut[r * LDSO_STRIDE + ch];
                float y = (xv - mean) * rstd * gamma[ch] + beta[ch];
                y = fmaxf(y, 0.0f);
                hout[((size_t)by * L + t0 + r) * DD + ch] = (half_t)y;
            }
        }
    }
}

// ---------------------------------------------------------------------------
extern "C" void kernel_launch(void* const* d_in, const int* in_sizes, int n_in,
                              void* d_out, int out_size, void* d_ws, size_t ws_size,
                              hipStream_t stream) {
    auto P = [&](int i) { return (const float*)d_in[i]; };
    const float* x        = (const float*)d_in[0];
    const int*   dur_t    = (const int*)d_in[1];
    const float* pitch_t  = (const float*)d_in[3];
    const float* energy_t = (const float*)d_in[4];
    const float* pitch_bins  = P(35);
    const float* energy_bins = P(36);
    const float* pitch_emb   = P(37);
    const float* energy_emb  = P(38);

    float* out         = (float*)d_out;
    float* xe          = out;                              // [B,T,D]
    float* log_dur     = out + (size_t)BB * TT * DD;       // [B,S]
    float* pitch_pred  = log_dur + (size_t)BB * SS;        // [B,T]
    float* energy_pred = pitch_pred + (size_t)BB * TT;     // [B,T]
    float* mel_len     = energy_pred + (size_t)BB * TT;    // [B]

    const size_t WPE = (size_t)KDIM * 256;                 // halves per packed layer
    half_t* wp = (half_t*)d_ws;
    int*    cum = (int*)((char*)d_ws + 6 * WPE * sizeof(half_t));
    half_t* h1  = (half_t*)((char*)cum + (size_t)BB * SS * sizeof(int));

    // Pack the six conv weights (dur1,dur2,pitch1,pitch2,energy1,energy2).
    const int widx[6] = {5, 9, 15, 19, 25, 29};
    for (int i = 0; i < 6; ++i)
        pack_w_kernel<<<KDIM, 256, 0, stream>>>(P(widx[i]), wp + (size_t)i * WPE);

    cumsum_kernel<<<1, 32, 0, stream>>>(dur_t, cum, mel_len);

    // Duration predictor on x (L = S).
    for (int b0 = 0; b0 < BB; b0 += CHUNK) {
        conv_kernel<float, false><<<dim3(SS / MT, CHUNK), 256, 0, stream>>>(
            x + (size_t)b0 * SS * DD, wp + 0 * WPE, P(6), P(7), P(8),
            nullptr, nullptr, h1, nullptr, SS);
        conv_kernel<half_t, true><<<dim3(SS / MT, CHUNK), 256, 0, stream>>>(
            h1, wp + 1 * WPE, P(10), P(11), P(12), P(13), P(14),
            nullptr, log_dur + (size_t)b0 * SS, SS);
    }

    // Length regulation -> xe (written fully each call, incl. masked zeros).
    regulate_kernel<<<dim3(TT / 8, BB), 256, 0, stream>>>(x, cum, xe);

    // Pitch predictor on xe (L = T).
    for (int b0 = 0; b0 < BB; b0 += CHUNK) {
        conv_kernel<float, false><<<dim3(TT / MT, CHUNK), 256, 0, stream>>>(
            xe + (size_t)b0 * TT * DD, wp + 2 * WPE, P(16), P(17), P(18),
            nullptr, nullptr, h1, nullptr, TT);
        conv_kernel<half_t, true><<<dim3(TT / MT, CHUNK), 256, 0, stream>>>(
            h1, wp + 3 * WPE, P(20), P(21), P(22), P(23), P(24),
            nullptr, pitch_pred + (size_t)b0 * TT, TT);
    }
    addemb_kernel<<<dim3(TT / 8, BB), 256, 0, stream>>>(pitch_t, pitch_bins, pitch_emb, xe);

    // Energy predictor on updated xe.
    for (int b0 = 0; b0 < BB; b0 += CHUNK) {
        conv_kernel<float, false><<<dim3(TT / MT, CHUNK), 256, 0, stream>>>(
            xe + (size_t)b0 * TT * DD, wp + 4 * WPE, P(26), P(27), P(28),
            nullptr, nullptr, h1, nullptr, TT);
        conv_kernel<half_t, true><<<dim3(TT / MT, CHUNK), 256, 0, stream>>>(
            h1, wp + 5 * WPE, P(30), P(31), P(32), P(33), P(34),
            nullptr, energy_pred + (size_t)b0 * TT, TT);
    }
    addemb_kernel<<<dim3(TT / 8, BB), 256, 0, stream>>>(energy_t, energy_bins, energy_emb, xe);
}
